// BModel_40312563040985
// MI455X (gfx1250) — compile-verified, matrix operand order checked by reading, not run on hardware
//
#include <hip/hip_runtime.h>

typedef __attribute__((ext_vector_type(4)))  float   v4f;
typedef __attribute__((ext_vector_type(8)))  float   v8f;
typedef __attribute__((ext_vector_type(8)))  __bf16  v8bf;
typedef __attribute__((ext_vector_type(16))) __bf16  v16bf;

#define IN_DIM  32768
#define CLASSES 100
#define NPAD    112           // 7 WMMA N-tiles of 16
#define NTILES  7
#define BATCH   4096
#define KSPLIT  32            // K-segments for occupancy
#define KSEG    (IN_DIM / KSPLIT)   // 1024 elements per segment

// ---------------- zero the output (atomics accumulate into it) ----------------
__global__ void zero_out_kernel(float* __restrict__ out, int n) {
    int i = blockIdx.x * blockDim.x + threadIdx.x;
    if (i < n) out[i] = 0.0f;
}

// ---------------- binarize W (fp32 -> bf16 sign), pad classes to 112 ----------
__global__ void binarize_kernel(const float* __restrict__ W, __bf16* __restrict__ bw, int total) {
    int idx = blockIdx.x * blockDim.x + threadIdx.x;
    if (idx >= total) return;
    int c = idx / IN_DIM;
    int k = idx - c * IN_DIM;
    float v = 0.0f;
    if (c < CLASSES) {
        float w = W[(size_t)c * IN_DIM + k];
        v = (w > 0.0f) ? 1.0f : ((w < 0.0f) ? -1.0f : 0.0f);
    }
    bw[idx] = (__bf16)v;
}

// ---------------- WMMA GEMM: out += (x * signW^T) * scale --------------------
// Wave owns 16 M-rows x 112 N-cols over one K-segment of 1024.
__global__ __launch_bounds__(128) void bnn_gemm_kernel(const float*  __restrict__ x,
                                                       const __bf16* __restrict__ bw,
                                                       float*        __restrict__ out) {
    const int lane      = threadIdx.x & 31;
    const int waveInBlk = threadIdx.x >> 5;
    const int w         = blockIdx.x * (blockDim.x >> 5) + waveInBlk;

    const int mTile = w % (BATCH / 16);       // 0..255
    const int kSeg  = w / (BATCH / 16);       // 0..31
    const int m0    = mTile * 16;

    const int laneMod = lane & 15;            // row (A) / class col (B)
    const int laneDiv = lane >> 4;            // which K-half this lane holds

    const float* xrow = x + (size_t)(m0 + laneMod) * IN_DIM;

    v8f acc[NTILES] = {};                     // 7 x 16x16 f32 accumulators

    const int kBeg = kSeg * KSEG;
    const int kEnd = kBeg + KSEG;

    for (int kb = kBeg; kb < kEnd; kb += 32) {
        const int kc0 = kb + laneDiv * 8;         // K chunk 0 (8 elems)
        const int kc1 = kb + 16 + laneDiv * 8;    // K chunk 1 (8 elems)

        // ---- A fragment: 16 fp32 -> 16 bf16 (ISA 16-bit A 16x32 layout) ----
        v4f a0 = *(const v4f*)(xrow + kc0);
        v4f a1 = *(const v4f*)(xrow + kc0 + 4);
        v4f a2 = *(const v4f*)(xrow + kc1);
        v4f a3 = *(const v4f*)(xrow + kc1 + 4);
        v16bf afrag;
#pragma unroll
        for (int i = 0; i < 4; ++i) {
            afrag[i]      = (__bf16)a0[i];
            afrag[4 + i]  = (__bf16)a1[i];
            afrag[8 + i]  = (__bf16)a2[i];
            afrag[12 + i] = (__bf16)a3[i];
        }

        // ---- 7 N-tiles: B columns are contiguous rows of binarized W ----
#pragma unroll
        for (int t = 0; t < NTILES; ++t) {
            const __bf16* bp = bw + (size_t)(t * 16 + laneMod) * IN_DIM;
            v8bf blo = *(const v8bf*)(bp + kc0);   // one b128 load
            v8bf bhi = *(const v8bf*)(bp + kc1);   // one b128 load
            v16bf bfrag = __builtin_shufflevector(blo, bhi,
                0, 1, 2, 3, 4, 5, 6, 7, 8, 9, 10, 11, 12, 13, 14, 15);
            acc[t] = __builtin_amdgcn_wmma_f32_16x16x32_bf16(
                /*neg_a=*/false, afrag, /*neg_b=*/false, bfrag,
                /*c_mod=*/(short)0, acc[t], /*reuse_a=*/false, /*reuse_b=*/false);
        }
    }

    // ---- epilogue: scale and atomically reduce K-split partials ----
    const float scale = 0.005524271728019903f;    // 1/sqrt(32768)
#pragma unroll
    for (int t = 0; t < NTILES; ++t) {
        int c = t * 16 + laneMod;
        if (c < CLASSES) {
#pragma unroll
            for (int r = 0; r < 8; ++r) {
                int m = m0 + r + 8 * laneDiv;     // C/D layout: VGPR r -> row r + 8*(lane/16)
                atomicAdd(out + (size_t)m * CLASSES + c, acc[t][r] * scale);
            }
        }
    }
}

extern "C" void kernel_launch(void* const* d_in, const int* in_sizes, int n_in,
                              void* d_out, int out_size, void* d_ws, size_t ws_size,
                              hipStream_t stream) {
    const float* x = (const float*)d_in[0];   // [4096, 32768] fp32
    const float* W = (const float*)d_in[1];   // [100, 32768] fp32
    float* out = (float*)d_out;               // [4096, 100] fp32
    __bf16* bw = (__bf16*)d_ws;               // [112, 32768] bf16 (7.3 MB scratch)

    // 1) zero output (K-split partials accumulate via atomics)
    zero_out_kernel<<<(out_size + 255) / 256, 256, 0, stream>>>(out, out_size);

    // 2) binarize + pad W
    int total = NPAD * IN_DIM;
    binarize_kernel<<<(total + 255) / 256, 256, 0, stream>>>(W, bw, total);

    // 3) WMMA GEMM: 8192 waves = 2048 blocks x 4 waves
    int totalWaves = (BATCH / 16) * KSPLIT;
    bnn_gemm_kernel<<<dim3(totalWaves / 4), dim3(128), 0, stream>>>(x, bw, out);
}
